// AudioLSTM_1666447310937
// MI455X (gfx1250) — compile-verified, matrix operand order checked by reading, not run on hardware
//
#include <hip/hip_runtime.h>

// ---------------------------------------------------------------------------
// Fused 2-layer LSTM + MLP head for MI455X (gfx1250, wave32, WMMA).
// One workgroup (256 threads = 8 waves) owns a 16-row batch tile and runs the
// full T=1000 recurrence for both LSTM layers. The two layers are software-
// pipelined one timestep apart so each iteration is exactly:
//   [ WMMA2(t) + WMMA1(t+1) ] barrier [ elem2(t) + elem1(t+1) + stage x ] barrier
// h/c state lives in LDS; all weights live in VGPRs as WMMA B-fragments;
// gate nonlinearities use gfx1250 hardware v_tanh_f32.
// ---------------------------------------------------------------------------

typedef __attribute__((ext_vector_type(16))) _Float16 v16h;
typedef __attribute__((ext_vector_type(8)))  float    v8f;

#define IN_F    26
#define H1      64
#define H2      32
#define F1      16
#define OUT_F   10
#define T_STEPS 1000
#define BATCH   512
#define KP      96      // padded K for both concatenated A panels

union FragH { v16h h; unsigned int u[8]; _Float16 e[16]; };

// ---- fast transcendentals: gfx1250 hardware v_tanh_f32 (probe-verified) ----
__device__ __forceinline__ float fast_rcp(float x) {
#if __has_builtin(__builtin_amdgcn_rcpf)
    return __builtin_amdgcn_rcpf(x);
#else
    return 1.0f / x;
#endif
}

#if __has_builtin(__builtin_amdgcn_tanhf)
__device__ __forceinline__ float tanh_hw(float x) { return __builtin_amdgcn_tanhf(x); }
#elif __has_builtin(__builtin_amdgcn_tanh_f32)
__device__ __forceinline__ float tanh_hw(float x) { return __builtin_amdgcn_tanh_f32(x); }
#else
__device__ __forceinline__ float tanh_hw(float x) {
    return 1.0f - 2.0f * fast_rcp(1.0f + __expf(2.0f * x));
}
#endif

__device__ __forceinline__ float sigf(float x) {
    // sigmoid(x) = 0.5*(1 + tanh(x/2))
    return __builtin_fmaf(0.5f, tanh_hw(0.5f * x), 0.5f);
}

__device__ __forceinline__ v8f wmma_acc(v16h a, v16h b, v8f c) {
    return __builtin_amdgcn_wmma_f32_16x16x32_f16(
        /*neg_a=*/false, a, /*neg_b=*/false, b,
        /*c_mod=*/(short)0, c, /*reuse_a=*/false, /*reuse_b=*/false);
}

// K index for dword pair p of a 16-bit WMMA A/B fragment (wave32 layout):
// lanes 0-15 cover K {0-7, 16-23}, lanes 16-31 cover K {8-15, 24-31}.
__device__ __forceinline__ int kmap_pair(int p, int hi8) {
    return ((p & 3) << 1) + ((p >> 2) << 4) + hi8;
}

__global__ __launch_bounds__(256)
void lstm_fused_kernel(const float* __restrict__ x,
                       const float* __restrict__ w_ih1, const float* __restrict__ w_hh1,
                       const float* __restrict__ b_ih1, const float* __restrict__ b_hh1,
                       const float* __restrict__ w_ih2, const float* __restrict__ w_hh2,
                       const float* __restrict__ b_ih2, const float* __restrict__ b_hh2,
                       const float* __restrict__ w_fc1, const float* __restrict__ b_fc1,
                       const float* __restrict__ w_fc2, const float* __restrict__ b_fc2,
                       float* __restrict__ out)
{
    __shared__ __align__(16) _Float16 A1[16 * KP];   // [x_t(26) | h1(64) | pad]
    __shared__ __align__(16) _Float16 A2[16 * KP];   // [h1(64)  | h2(32)]
    __shared__ float G1[16 * 256];                   // raw layer-1 gates
    __shared__ float G2[16 * 128];                   // raw layer-2 gates
    __shared__ float C1s[16 * H1];
    __shared__ float C2s[16 * H2];
    __shared__ float H2F[16 * H2];                   // final h2 (f32) for FC head

    const int tid  = threadIdx.x;
    const int lane = tid & 31;
    const int wave = tid >> 5;
    const int b0   = blockIdx.x * 16;                // batch-row base of this WG

    const int col  = lane & 15;
    const int hi8  = (lane >> 4) << 3;
    const int m    = lane & 15;

    // ---- zero-init persistent LDS state (h=0, c=0, pads) ----
    for (int i = tid; i < 16 * KP; i += 256) {
        A1[i] = (_Float16)0.0f;
        A2[i] = (_Float16)0.0f;
    }
    for (int i = tid; i < 16 * H1; i += 256) C1s[i] = 0.0f;
    for (int i = tid; i < 16 * H2; i += 256) C2s[i] = 0.0f;

    // ---- persistent weight B-fragments in VGPRs ----
    // Layer1: wave w owns gate-column tiles {w, w+8} (256 cols).
    // Layer2: wave w owns gate-column tile w (128 cols).
    v16h B1f[6];
    v16h B2f[3];
    float bias1a, bias1b, bias2;
    {
        const int n0 = 16 * wave + col;
        const int n1 = n0 + 128;
        const int n2 = 16 * wave + col;
        bias1a = b_ih1[n0] + b_hh1[n0];
        bias1b = b_ih1[n1] + b_hh1[n1];
        bias2  = b_ih2[n2] + b_hh2[n2];
        #pragma unroll
        for (int kc = 0; kc < 3; ++kc) {
            FragH fa, fb, f2;
            #pragma unroll
            for (int ee = 0; ee < 16; ++ee) {
                const int p = ee >> 1, j = ee & 1;
                const int k = kc * 32 + kmap_pair(p, hi8) + j;
                float wv0 = 0.0f, wv1 = 0.0f;
                if (k < IN_F) {
                    wv0 = w_ih1[n0 * IN_F + k];
                    wv1 = w_ih1[n1 * IN_F + k];
                } else if (k < IN_F + H1) {
                    wv0 = w_hh1[n0 * H1 + (k - IN_F)];
                    wv1 = w_hh1[n1 * H1 + (k - IN_F)];
                }
                fa.e[ee] = (_Float16)wv0;
                fb.e[ee] = (_Float16)wv1;
                const float wv2 = (k < H1) ? w_ih2[n2 * H1 + k]
                                           : w_hh2[n2 * H2 + (k - H1)];
                f2.e[ee] = (_Float16)wv2;
            }
            B1f[kc]     = fa.h;
            B1f[3 + kc] = fb.h;
            B2f[kc]     = f2.h;
        }
    }

    // ---- phase helpers ----
    auto stage_x = [&](int t) {
        for (int i = tid; i < 16 * IN_F; i += 256) {
            const int r = i / IN_F, f = i % IN_F;
            const size_t base = (size_t)(b0 + r) * (IN_F * T_STEPS) + (size_t)f * T_STEPS;
            A1[r * KP + f] = (_Float16)x[base + t];
            if (((t & 3) == 0) && (t + 16 < T_STEPS))
                __builtin_prefetch(&x[base + t + 16], 0, 0);   // global_prefetch_b8
        }
    };

    auto wmma1 = [&]() {   // G1 = A1[16x96] @ W1cat[96x256] + bias
        v16h aF[3];
        #pragma unroll
        for (int kc = 0; kc < 3; ++kc) {
            FragH fr;
            #pragma unroll
            for (int p = 0; p < 8; ++p)
                fr.u[p] = *(const unsigned int*)(A1 + m * KP + kc * 32 + kmap_pair(p, hi8));
            aF[kc] = fr.h;
        }
        v8f acc0, acc1;
        #pragma unroll
        for (int v = 0; v < 8; ++v) { acc0[v] = bias1a; acc1[v] = bias1b; }
        #pragma unroll
        for (int kc = 0; kc < 3; ++kc) {
            acc0 = wmma_acc(aF[kc], B1f[kc],     acc0);
            acc1 = wmma_acc(aF[kc], B1f[3 + kc], acc1);
        }
        const int cb = 16 * wave + col;
        #pragma unroll
        for (int v = 0; v < 8; ++v) {
            G1[(hi8 + v) * 256 + cb]       = acc0[v];
            G1[(hi8 + v) * 256 + cb + 128] = acc1[v];
        }
    };

    auto wmma2 = [&]() {   // G2 = A2[16x96] @ W2cat[96x128] + bias
        v16h aF[3];
        #pragma unroll
        for (int kc = 0; kc < 3; ++kc) {
            FragH fr;
            #pragma unroll
            for (int p = 0; p < 8; ++p)
                fr.u[p] = *(const unsigned int*)(A2 + m * KP + kc * 32 + kmap_pair(p, hi8));
            aF[kc] = fr.h;
        }
        v8f acc2;
        #pragma unroll
        for (int v = 0; v < 8; ++v) acc2[v] = bias2;
        #pragma unroll
        for (int kc = 0; kc < 3; ++kc)
            acc2 = wmma_acc(aF[kc], B2f[kc], acc2);
        const int cb = 16 * wave + col;
        #pragma unroll
        for (int v = 0; v < 8; ++v)
            G2[(hi8 + v) * 128 + cb] = acc2[v];
    };

    auto elem1 = [&]() {   // LSTM-1 cell update: 16x64 elements
        #pragma unroll
        for (int q = 0; q < 4; ++q) {
            const int idx = tid * 4 + q;
            const int mm = idx >> 6, j = idx & 63;
            const float iv = sigf   (G1[mm * 256 +       j]);
            const float fv = sigf   (G1[mm * 256 +  64 + j]);
            const float gv = tanh_hw(G1[mm * 256 + 128 + j]);
            const float ov = sigf   (G1[mm * 256 + 192 + j]);
            const float cc = fv * C1s[mm * H1 + j] + iv * gv;
            C1s[mm * H1 + j] = cc;
            const float hh = ov * tanh_hw(cc);
            A1[mm * KP + IN_F + j] = (_Float16)hh;   // layer-1 recurrence input
            A2[mm * KP + j]        = (_Float16)hh;   // layer-2 input
        }
    };

    auto elem2 = [&]() {   // LSTM-2 cell update: 16x32 elements
        #pragma unroll
        for (int q = 0; q < 2; ++q) {
            const int idx = tid * 2 + q;
            const int mm = idx >> 5, j = idx & 31;
            const float iv = sigf   (G2[mm * 128 +      j]);
            const float fv = sigf   (G2[mm * 128 + 32 + j]);
            const float gv = tanh_hw(G2[mm * 128 + 64 + j]);
            const float ov = sigf   (G2[mm * 128 + 96 + j]);
            const float cc = fv * C2s[mm * H2 + j] + iv * gv;
            C2s[mm * H2 + j] = cc;
            const float hh = ov * tanh_hw(cc);
            A2[mm * KP + H1 + j] = (_Float16)hh;     // layer-2 recurrence input
            H2F[mm * H2 + j]     = hh;
        }
    };

    // ---- prologue: fill the 1-step pipeline ----
    __syncthreads();            // zero-init complete before x staging overlaps it
    stage_x(0);
    __syncthreads();
    wmma1();                    // G1(t=0)
    __syncthreads();
    elem1();                    // h1_0
    stage_x(1);
    __syncthreads();

    // ---- steady state: 2 barriers per timestep ----
    for (int t = 0; t < T_STEPS; ++t) {
        wmma2();                              // G2(t)   from h1_t, h2_{t-1}
        if (t + 1 < T_STEPS) wmma1();         // G1(t+1) from x_{t+1}, h1_t
        __syncthreads();
        elem2();                              // h2_t
        if (t + 1 < T_STEPS) {
            elem1();                          // h1_{t+1}
            if (t + 2 < T_STEPS) stage_x(t + 2);
        }
        __syncthreads();
    }

    // ---- MLP head on final h2 ----
    if (tid < 16 * OUT_F) {
        const int mm = tid / OUT_F, o = tid % OUT_F;
        float acc = b_fc2[o];
        #pragma unroll
        for (int j = 0; j < F1; ++j) {
            float s = b_fc1[j];
            #pragma unroll
            for (int k = 0; k < H2; ++k)
                s += H2F[mm * H2 + k] * w_fc1[j * H2 + k];
            s = fmaxf(s, 0.0f);
            acc += s * w_fc2[o * F1 + j];
        }
        out[(b0 + mm) * OUT_F + o] = acc;
    }
}

extern "C" void kernel_launch(void* const* d_in, const int* in_sizes, int n_in,
                              void* d_out, int out_size, void* d_ws, size_t ws_size,
                              hipStream_t stream) {
    (void)in_sizes; (void)n_in; (void)d_ws; (void)ws_size; (void)out_size;
    const float* x     = (const float*)d_in[0];
    const float* w_ih1 = (const float*)d_in[1];
    const float* w_hh1 = (const float*)d_in[2];
    const float* b_ih1 = (const float*)d_in[3];
    const float* b_hh1 = (const float*)d_in[4];
    const float* w_ih2 = (const float*)d_in[5];
    const float* w_hh2 = (const float*)d_in[6];
    const float* b_ih2 = (const float*)d_in[7];
    const float* b_hh2 = (const float*)d_in[8];
    const float* w_fc1 = (const float*)d_in[9];
    const float* b_fc1 = (const float*)d_in[10];
    const float* w_fc2 = (const float*)d_in[11];
    const float* b_fc2 = (const float*)d_in[12];
    float* out = (float*)d_out;

    dim3 grid(BATCH / 16);   // 32 workgroups, one 16-row batch tile each
    dim3 block(256);         // 8 waves of 32
    hipLaunchKernelGGL(lstm_fused_kernel, grid, block, 0, stream,
                       x, w_ih1, w_hh1, b_ih1, b_hh1,
                       w_ih2, w_hh2, b_ih2, b_hh2,
                       w_fc1, b_fc1, w_fc2, b_fc2, out);
}
